// VisionTransformerPacked_53575422051117
// MI455X (gfx1250) — matrix operand order repaired and checked by reading, not compiled, not run on hardware
//
#include <hip/hip_runtime.h>
#include <hip/hip_bf16.h>

// ---------------- model constants ----------------
#define Bb   4
#define Ss   1024
#define Dd   768
#define Hh   12
#define Ll   4
#define Mm   4
#define HDd  64
#define DFf  3072

typedef __attribute__((ext_vector_type(16))) __bf16 v16bf;
typedef __attribute__((ext_vector_type(8)))  float  v8f;
typedef __attribute__((ext_vector_type(4)))  unsigned int v4u;
typedef __attribute__((ext_vector_type(8)))  int v8i;
typedef __attribute__((ext_vector_type(4)))  int v4i;

union Frag16 { v16bf v; uint4 q[2]; __bf16 h[16]; };
union U4B    { uint4 u; __bf16 h[8]; };

#define WMMA_BF16(a, b, c) \
  __builtin_amdgcn_wmma_f32_16x16x32_bf16(false, (a), false, (b), (short)0, (c), false, false)

// ---------------- TDM: DMA a [rows=128 x k=32] bf16 tile (row stride K elems) to LDS ----
// Descriptor per CDNA5 ISA ch.8: group0 {count,lds_addr,global_addr,type=2},
// group1 {data_size=2B, tensor dims/strides, tile 32x128}. 2-D tile, groups 2/3 zero.
__device__ __forceinline__ void tdm_load_A_tile(const __bf16* gptr, unsigned ldsOff, int K) {
  unsigned long long ga = (unsigned long long)(size_t)gptr;
  v4u g0;
  g0[0] = 1u;                                       // count=1, user descriptor
  g0[1] = ldsOff;                                   // lds_addr (bytes)
  g0[2] = (unsigned)(ga & 0xFFFFFFFFu);             // global_addr[31:0]
  g0[3] = (unsigned)((ga >> 32) & 0x01FFFFFFu) | (2u << 30); // global_addr[56:32] | type=2
  v8i g1;
  g1[0] = (int)(1u << 16);                          // wg_mask=0, data_size=1 (2 bytes)
  g1[1] = (int)(((unsigned)K & 0xFFFFu) << 16);     // tensor_dim0[15:0] @ bits 63:48
  g1[2] = (int)((((unsigned)K >> 16) & 0xFFFFu) | (4096u << 16)); // dim0 hi | tensor_dim1 lo
  g1[3] = (int)(32u << 16);                         // tensor_dim1 hi=0 | tile_dim0=32
  g1[4] = (int)(128u);                              // tile_dim1=128, tile_dim2=0
  g1[5] = (int)(unsigned)K;                         // tensor_dim0_stride[31:0]
  g1[6] = 0;                                        // stride hi | dim1_stride lo
  g1[7] = 0;
  v4i z4 = {0, 0, 0, 0};
  v8i z8 = {0, 0, 0, 0, 0, 0, 0, 0};
  __builtin_amdgcn_tensor_load_to_lds(g0, g1, z4, z4, z8, 0);
}

// ---------------- fp32 -> bf16 convert ----------------
__global__ __launch_bounds__(256) void k_f32_to_bf16(const float* __restrict__ in,
                                                     __bf16* __restrict__ out, int n) {
  int i = blockIdx.x * 256 + threadIdx.x;
  if (i < n) out[i] = (__bf16)in[i];
}

// ---------------- positional-embedding gather ----------------
__global__ __launch_bounds__(256) void k_pos_gather(const float* __restrict__ pos_table,
                                                    const int* __restrict__ pos_ids,
                                                    float* __restrict__ out, int n) {
  int i = blockIdx.x * 256 + threadIdx.x;
  if (i < n) {
    int row = i / Dd;
    int c   = i - row * Dd;
    out[i] = pos_table[(size_t)pos_ids[row] * Dd + c];
  }
}

// ---------------- LayerNorm (row = one block), bf16 output ----------------
__global__ __launch_bounds__(256) void k_layernorm_bf16(const float* __restrict__ x,
                                                        const float* __restrict__ g,
                                                        const float* __restrict__ bta,
                                                        __bf16* __restrict__ out) {
  __shared__ float redS[8], redS2[8];
  int row = blockIdx.x, tid = threadIdx.x;
  const float* xr = x + (size_t)row * Dd;
  float v0 = xr[tid], v1 = xr[tid + 256], v2 = xr[tid + 512];
  float s = v0 + v1 + v2, s2 = v0 * v0 + v1 * v1 + v2 * v2;
  for (int k = 16; k >= 1; k >>= 1) { s += __shfl_xor(s, k); s2 += __shfl_xor(s2, k); }
  if ((tid & 31) == 0) { redS[tid >> 5] = s; redS2[tid >> 5] = s2; }
  __syncthreads();
  float ts = 0.f, ts2 = 0.f;
#pragma unroll
  for (int i = 0; i < 8; i++) { ts += redS[i]; ts2 += redS2[i]; }
  float mean = ts * (1.f / Dd);
  float var  = ts2 * (1.f / Dd) - mean * mean;
  float inv  = rsqrtf(var + 1e-5f);
  __bf16* orow = out + (size_t)row * Dd;
  orow[tid]       = (__bf16)((v0 - mean) * inv * g[tid]       + bta[tid]);
  orow[tid + 256] = (__bf16)((v1 - mean) * inv * g[tid + 256] + bta[tid + 256]);
  orow[tid + 512] = (__bf16)((v2 - mean) * inv * g[tid + 512] + bta[tid + 512]);
}

// ---------------- tiled bf16 WMMA GEMM (double-buffered, TDM A-tiles) ----------------
// C[M,N] = A[M,K] @ B[K,N] (+bias) (+gelu) (+residual), fp32 and/or bf16 out.
// M,N multiples of 128; K multiple of 32.
#define EP_GELU 1
#define EP_RES  2
__global__ __launch_bounds__(256) void k_gemm_bf16(
    const __bf16* __restrict__ A, const __bf16* __restrict__ Bm,
    const float* __restrict__ bias, const float* __restrict__ resid,
    float* __restrict__ outF, __bf16* __restrict__ outB,
    int M, int N, int K, int flags) {
  __shared__ __align__(16) __bf16 As[2][128][32];   // [buf][m][k]   (TDM destination)
  __shared__ __align__(16) __bf16 Bt[2][128][40];   // [buf][n][k]   padded, 16B-aligned rows
  int tid = threadIdx.x, lane = tid & 31, wid = tid >> 5, half = lane >> 4;
  int nBase = blockIdx.x * 128, mBase = blockIdx.y * 128;
  int wm = (wid >> 2) * 64, wn = (wid & 3) * 32;

  v8f acc[4][2];
#pragma unroll
  for (int i = 0; i < 4; i++)
#pragma unroll
    for (int j = 0; j < 2; j++)
#pragma unroll
      for (int v = 0; v < 8; v++) acc[i][j][v] = 0.f;

  int bk = tid & 31, bns = (tid >> 5) * 16;
  int kTiles = K >> 5;
  unsigned asOff[2] = { (unsigned)(size_t)&As[0][0][0], (unsigned)(size_t)&As[1][0][0] };

  // ---- prologue: stage tile 0 ----
  if (wid == 0) tdm_load_A_tile(A + (size_t)mBase * K, asOff[0], K);
  {
    const __bf16* bp = Bm + (size_t)bk * N + nBase + bns;
    U4B b0, b1; b0.u = *(const uint4*)bp; b1.u = *(const uint4*)(bp + 8);
#pragma unroll
    for (int j = 0; j < 8; j++) { Bt[0][bns + j][bk] = b0.h[j]; Bt[0][bns + 8 + j][bk] = b1.h[j]; }
  }
  if (wid == 0) __builtin_amdgcn_s_wait_tensorcnt(0);

  for (int kt = 0; kt < kTiles; ++kt) {
    int cur = kt & 1, nxt = cur ^ 1;
    __syncthreads();                    // tile kt staged; buffers nxt free to overwrite
    bool more = (kt + 1) < kTiles;
    U4B nb0, nb1;
    if (more) {
      if (wid == 0)
        tdm_load_A_tile(A + (size_t)mBase * K + (kt + 1) * 32, asOff[nxt], K);
      const __bf16* bp = Bm + (size_t)((kt + 1) * 32 + bk) * N + nBase + bns;
      nb0.u = *(const uint4*)bp; nb1.u = *(const uint4*)(bp + 8);
      if (kt + 2 < kTiles)
        __builtin_prefetch(Bm + (size_t)((kt + 2) * 32 + bk) * N + nBase + bns, 0, 1);
    }

    Frag16 af[4], bf[2];
#pragma unroll
    for (int i = 0; i < 4; i++) {
      int r = wm + i * 16 + (lane & 15);
      af[i].q[0] = *(const uint4*)&As[cur][r][half * 8];
      af[i].q[1] = *(const uint4*)&As[cur][r][16 + half * 8];
    }
#pragma unroll
    for (int j = 0; j < 2; j++) {
      int n = wn + j * 16 + (lane & 15);
      bf[j].q[0] = *(const uint4*)&Bt[cur][n][half * 8];
      bf[j].q[1] = *(const uint4*)&Bt[cur][n][16 + half * 8];
    }
#pragma unroll
    for (int i = 0; i < 4; i++)
#pragma unroll
      for (int j = 0; j < 2; j++)
        acc[i][j] = WMMA_BF16(af[i].v, bf[j].v, acc[i][j]);

    if (more) {
#pragma unroll
      for (int j = 0; j < 8; j++) {
        Bt[nxt][bns + j][bk]     = nb0.h[j];
        Bt[nxt][bns + 8 + j][bk] = nb1.h[j];
      }
      if (wid == 0) __builtin_amdgcn_s_wait_tensorcnt(0);
    }
  }

  // ---- epilogue ----
#pragma unroll
  for (int i = 0; i < 4; i++) {
#pragma unroll
    for (int j = 0; j < 2; j++) {
      int col = nBase + wn + j * 16 + (lane & 15);
      float bv = bias ? bias[col] : 0.f;
#pragma unroll
      for (int v = 0; v < 8; v++) {
        int row = mBase + wm + i * 16 + half * 8 + v;
        float val = acc[i][j][v] + bv;
        if (flags & EP_GELU) val = 0.5f * val * (1.f + erff(val * 0.70710678118f));
        size_t idx = (size_t)row * N + col;
        if (flags & EP_RES) val += resid[idx];
        if (outF) outF[idx] = val;
        if (outB) outB[idx] = (__bf16)val;
      }
    }
  }
}

// ---------------- flash attention (bf16 WMMA, packed block-diag mask) ----------------
// qkv: [B,S,3*D] bf16, row col = e*768 + h*64 + d.  o: [B,S,D] bf16.
__global__ __launch_bounds__(256) void k_attention(const __bf16* __restrict__ qkv,
                                                   const int* __restrict__ seq_ids,
                                                   __bf16* __restrict__ o) {
  __shared__ __align__(16) __bf16 Kt[32][64];    // [key][d]
  __shared__ __align__(16) __bf16 Vt[64][40];    // [d][key], padded
  __shared__ __align__(16) __bf16 Pl[8][16][32]; // per-wave P scratch [row][key]
  __shared__ int sidK[32];
  int tid = threadIdx.x, lane = tid & 31, wid = tid >> 5, half = lane >> 4;
  int qTile = blockIdx.x * 128, h = blockIdx.y, b = blockIdx.z;
  int qRow0 = qTile + wid * 16;
  const int rowStride = 3 * Dd;

  Frag16 qf[2];
  {
    int r = qRow0 + (lane & 15);
    const __bf16* qp = qkv + (size_t)(b * Ss + r) * rowStride + h * HDd;
    qf[0].q[0] = *(const uint4*)(qp + half * 8);
    qf[0].q[1] = *(const uint4*)(qp + 16 + half * 8);
    qf[1].q[0] = *(const uint4*)(qp + 32 + half * 8);
    qf[1].q[1] = *(const uint4*)(qp + 48 + half * 8);
  }
  int sidQ[8];
#pragma unroll
  for (int v = 0; v < 8; v++) sidQ[v] = seq_ids[b * Ss + qRow0 + half * 8 + v];

  float rowm[8], rowl[8];
  v8f Oc[4];
#pragma unroll
  for (int v = 0; v < 8; v++) { rowm[v] = -3.0e38f; rowl[v] = 0.f; }
#pragma unroll
  for (int df = 0; df < 4; df++)
#pragma unroll
    for (int v = 0; v < 8; v++) Oc[df][v] = 0.f;

  int lkey = tid >> 3, ldc = (tid & 7) * 8;
  // prologue: stage key block 0
  U4B kreg, vreg; int sreg = 0;
  {
    const __bf16* kp = qkv + (size_t)(b * Ss + lkey) * rowStride + Dd + h * HDd + ldc;
    kreg.u = *(const uint4*)kp;
    vreg.u = *(const uint4*)(kp + Dd);
    if (tid < 32) sreg = seq_ids[b * Ss + tid];
  }
  *(uint4*)&Kt[lkey][ldc] = kreg.u;
#pragma unroll
  for (int j = 0; j < 8; j++) Vt[ldc + j][lkey] = vreg.h[j];
  if (tid < 32) sidK[tid] = sreg;
  __syncthreads();

  for (int kb = 0; kb < Ss / 32; ++kb) {
    bool more = (kb + 1) < Ss / 32;
    if (more) {  // issue next block's global loads before the matrix work
      const __bf16* kp = qkv + (size_t)(b * Ss + (kb + 1) * 32 + lkey) * rowStride + Dd + h * HDd + ldc;
      kreg.u = *(const uint4*)kp;
      vreg.u = *(const uint4*)(kp + Dd);
      if (tid < 32) sreg = seq_ids[b * Ss + (kb + 1) * 32 + tid];
    }

    float sc[2][8];
#pragma unroll
    for (int sub = 0; sub < 2; sub++) {
      Frag16 kf0, kf1;
      int kk = sub * 16 + (lane & 15);
      kf0.q[0] = *(const uint4*)&Kt[kk][half * 8];
      kf0.q[1] = *(const uint4*)&Kt[kk][16 + half * 8];
      kf1.q[0] = *(const uint4*)&Kt[kk][32 + half * 8];
      kf1.q[1] = *(const uint4*)&Kt[kk][48 + half * 8];
      v8f c;
#pragma unroll
      for (int v = 0; v < 8; v++) c[v] = 0.f;
      c = WMMA_BF16(qf[0].v, kf0.v, c);
      c = WMMA_BF16(qf[1].v, kf1.v, c);
      int sk = sidK[kk];
#pragma unroll
      for (int v = 0; v < 8; v++)
        sc[sub][v] = c[v] * 0.125f + ((sk == sidQ[v]) ? 0.f : -1e9f);
    }
    // online softmax over the 32 keys (reductions within 16-lane halves)
#pragma unroll
    for (int v = 0; v < 8; v++) {
      float mx = fmaxf(sc[0][v], sc[1][v]);
#pragma unroll
      for (int m = 1; m < 16; m <<= 1) mx = fmaxf(mx, __shfl_xor(mx, m));
      float mnew = fmaxf(rowm[v], mx);
      float r  = __expf(rowm[v] - mnew);
      float p0 = __expf(sc[0][v] - mnew);
      float p1 = __expf(sc[1][v] - mnew);
      float ps = p0 + p1;
#pragma unroll
      for (int m = 1; m < 16; m <<= 1) ps += __shfl_xor(ps, m);
      rowl[v] = rowl[v] * r + ps;
      rowm[v] = mnew;
#pragma unroll
      for (int df = 0; df < 4; df++) Oc[df][v] *= r;
      int prow = half * 8 + v;
      Pl[wid][prow][lane & 15]        = (__bf16)p0;
      Pl[wid][prow][16 + (lane & 15)] = (__bf16)p1;
    }
    // P @ V
    Frag16 af;
    {
      int r = lane & 15;
      af.q[0] = *(const uint4*)&Pl[wid][r][half * 8];
      af.q[1] = *(const uint4*)&Pl[wid][r][16 + half * 8];
    }
#pragma unroll
    for (int df = 0; df < 4; ++df) {
      Frag16 vf;
      int d = df * 16 + (lane & 15);
      vf.q[0] = *(const uint4*)&Vt[d][half * 8];
      vf.q[1] = *(const uint4*)&Vt[d][16 + half * 8];
      Oc[df] = WMMA_BF16(af.v, vf.v, Oc[df]);
    }

    if (more) {
      __syncthreads();                 // all waves done reading Kt/Vt
      *(uint4*)&Kt[lkey][ldc] = kreg.u;
#pragma unroll
      for (int j = 0; j < 8; j++) Vt[ldc + j][lkey] = vreg.h[j];
      if (tid < 32) sidK[tid] = sreg;
      __syncthreads();                 // next block staged
    }
  }
#pragma unroll
  for (int df = 0; df < 4; df++) {
    int col = h * HDd + df * 16 + (lane & 15);
#pragma unroll
    for (int v = 0; v < 8; v++) {
      int row = qRow0 + half * 8 + v;
      o[(size_t)(b * Ss + row) * Dd + col] = (__bf16)(Oc[df][v] / rowl[v]);
    }
  }
}

// ---------------- attention-pool head (tiny; plain VALU) ----------------
__global__ __launch_bounds__(256) void k_pool(const float* __restrict__ kv,
                                              const int* __restrict__ seq_ids,
                                              const float* __restrict__ latent,
                                              float* __restrict__ out) {
  __shared__ float sc[Ss];
  __shared__ int sid[Ss];
  __shared__ float qsh[HDd];
  __shared__ float redbuf[8];
  __shared__ float bmax[Mm], bsum[Mm];
  int tid = threadIdx.x, h = blockIdx.x, b = blockIdx.y;
  if (tid < HDd) qsh[tid] = latent[h * HDd + tid];
  __syncthreads();
  for (int s = tid; s < Ss; s += 256) {
    const float* kp = kv + (size_t)(b * Ss + s) * (2 * Dd) + h * HDd;
    float acc = 0.f;
    for (int d = 0; d < HDd; d++) acc += qsh[d] * kp[d];
    sc[s]  = acc * 0.125f;
    sid[s] = seq_ids[b * Ss + s];
  }
  __syncthreads();
  for (int m = 0; m < Mm; m++) {
    float pm = -3e38f;
    for (int s = tid; s < Ss; s += 256) if (sid[s] == m) pm = fmaxf(pm, sc[s]);
    for (int k = 16; k >= 1; k >>= 1) pm = fmaxf(pm, __shfl_xor(pm, k));
    if ((tid & 31) == 0) redbuf[tid >> 5] = pm;
    __syncthreads();
    if (tid == 0) { float t = -3e38f; for (int i = 0; i < 8; i++) t = fmaxf(t, redbuf[i]); bmax[m] = t; }
    __syncthreads();
    float ps = 0.f;
    for (int s = tid; s < Ss; s += 256) if (sid[s] == m) ps += __expf(sc[s] - bmax[m]);
    for (int k = 16; k >= 1; k >>= 1) ps += __shfl_xor(ps, k);
    if ((tid & 31) == 0) redbuf[tid >> 5] = ps;
    __syncthreads();
    if (tid == 0) { float t = 0.f; for (int i = 0; i < 8; i++) t += redbuf[i]; bsum[m] = t; }
    __syncthreads();
  }
  for (int s = tid; s < Ss; s += 256) {
    int m = sid[s];
    sc[s] = (m < Mm) ? __expf(sc[s] - bmax[m]) / bsum[m] : 0.f;
  }
  __syncthreads();
  int m = tid >> 6, d = tid & 63;
  float acc = 0.f;
  for (int s = 0; s < Ss; s++)
    if (sid[s] == m) acc += sc[s] * kv[(size_t)(b * Ss + s) * (2 * Dd) + Dd + h * HDd + d];
  out[(size_t)(b * Mm + m) * Dd + h * HDd + d] = acc;
}

// ---------------- orchestration ----------------
extern "C" void kernel_launch(void* const* d_in, const int* in_sizes, int n_in,
                              void* d_out, int out_size, void* d_ws, size_t ws_size,
                              hipStream_t stream) {
  const float* x         = (const float*)d_in[0];
  const int*   seq_ids   = (const int*)d_in[1];
  const int*   pos_ids   = (const int*)d_in[2];
  const float* w_patch   = (const float*)d_in[3];
  const float* b_patch   = (const float*)d_in[4];
  const float* pos_table = (const float*)d_in[5];
  const float* ln1_g = (const float*)d_in[6];
  const float* ln1_b = (const float*)d_in[7];
  const float* w_qkv = (const float*)d_in[8];
  const float* b_qkv = (const float*)d_in[9];
  const float* w_o   = (const float*)d_in[10];
  const float* b_o   = (const float*)d_in[11];
  const float* ln2_g = (const float*)d_in[12];
  const float* ln2_b = (const float*)d_in[13];
  const float* w_fc1 = (const float*)d_in[14];
  const float* b_fc1 = (const float*)d_in[15];
  const float* w_fc2 = (const float*)d_in[16];
  const float* b_fc2 = (const float*)d_in[17];
  const float* lnf_g = (const float*)d_in[18];
  const float* lnf_b = (const float*)d_in[19];
  const float* latent = (const float*)d_in[20];
  const float* w_kv  = (const float*)d_in[21];
  const float* b_kv  = (const float*)d_in[22];
  float* out = (float*)d_out;
  (void)in_sizes; (void)n_in; (void)out_size; (void)ws_size;

  const int ROWS = Bb * Ss; // 4096
  char* ws = (char*)d_ws;
  size_t off = 0;
  auto take = [&](size_t bytes) -> char* {
    char* p = ws + off;
    off = (off + bytes + 255) & ~(size_t)255;
    return p;
  };
  float*  h    = (float*)take((size_t)ROWS * Dd * 4);          // residual stream
  __bf16* y    = (__bf16*)take((size_t)ROWS * Dd * 2);         // LN output
  __bf16* qkvb = (__bf16*)take((size_t)ROWS * 3 * Dd * 2);     // qkv (aliased by fc1 act)
  __bf16* obuf = (__bf16*)take((size_t)ROWS * Dd * 2);         // attention out
  __bf16* abuf = qkvb;                                         // fc1 act reuses qkv+o region
  float*  f32t = (float*)take((size_t)ROWS * 2 * Dd * 4);      // pos-res / kv
  __bf16* wbuf = (__bf16*)take((size_t)Dd * DFf * 2);          // per-GEMM bf16 weights
  __bf16* xbf  = (__bf16*)take((size_t)ROWS * Dd * 2);         // bf16 x / final-LN out

  dim3 blk(256);
  auto cdiv = [](int a, int b) { return (a + b - 1) / b; };
  const int nXD = ROWS * Dd;

  // patch embed: h = x @ w_patch + b_patch + pos_table[pos_ids]
  k_f32_to_bf16<<<cdiv(nXD, 256), blk, 0, stream>>>(x, xbf, nXD);
  k_f32_to_bf16<<<cdiv(Dd * Dd, 256), blk, 0, stream>>>(w_patch, wbuf, Dd * Dd);
  k_pos_gather<<<cdiv(nXD, 256), blk, 0, stream>>>(pos_table, pos_ids, f32t, nXD);
  k_gemm_bf16<<<dim3(Dd / 128, ROWS / 128), blk, 0, stream>>>(
      xbf, wbuf, b_patch, f32t, h, (__bf16*)nullptr, ROWS, Dd, Dd, EP_RES);

  for (int l = 0; l < Ll; l++) {
    k_layernorm_bf16<<<ROWS, blk, 0, stream>>>(h, ln1_g + l * Dd, ln1_b + l * Dd, y);
    k_f32_to_bf16<<<cdiv(Dd * 3 * Dd, 256), blk, 0, stream>>>(w_qkv + (size_t)l * Dd * 3 * Dd, wbuf, Dd * 3 * Dd);
    k_gemm_bf16<<<dim3(3 * Dd / 128, ROWS / 128), blk, 0, stream>>>(
        y, wbuf, b_qkv + l * 3 * Dd, (const float*)nullptr, (float*)nullptr, qkvb, ROWS, 3 * Dd, Dd, 0);
    k_attention<<<dim3(Ss / 128, Hh, Bb), blk, 0, stream>>>(qkvb, seq_ids, obuf);
    k_f32_to_bf16<<<cdiv(Dd * Dd, 256), blk, 0, stream>>>(w_o + (size_t)l * Dd * Dd, wbuf, Dd * Dd);
    k_gemm_bf16<<<dim3(Dd / 128, ROWS / 128), blk, 0, stream>>>(
        obuf, wbuf, b_o + l * Dd, h, h, (__bf16*)nullptr, ROWS, Dd, Dd, EP_RES);
    k_layernorm_bf16<<<ROWS, blk, 0, stream>>>(h, ln2_g + l * Dd, ln2_b + l * Dd, y);
    k_f32_to_bf16<<<cdiv(Dd * DFf, 256), blk, 0, stream>>>(w_fc1 + (size_t)l * Dd * DFf, wbuf, Dd * DFf);
    k_gemm_bf16<<<dim3(DFf / 128, ROWS / 128), blk, 0, stream>>>(
        y, wbuf, b_fc1 + l * DFf, (const float*)nullptr, (float*)nullptr, abuf, ROWS, DFf, Dd, EP_GELU);
    k_f32_to_bf16<<<cdiv(DFf * Dd, 256), blk, 0, stream>>>(w_fc2 + (size_t)l * DFf * Dd, wbuf, DFf * Dd);
    k_gemm_bf16<<<dim3(Dd / 128, ROWS / 128), blk, 0, stream>>>(
        abuf, wbuf, b_fc2 + l * Dd, h, h, (__bf16*)nullptr, ROWS, Dd, DFf, EP_RES);
  }

  // final LN -> bf16, kv projection -> fp32, then pool head
  k_layernorm_bf16<<<ROWS, blk, 0, stream>>>(h, lnf_g, lnf_b, xbf);
  k_f32_to_bf16<<<cdiv(Dd * 2 * Dd, 256), blk, 0, stream>>>(w_kv, wbuf, Dd * 2 * Dd);
  k_gemm_bf16<<<dim3(2 * Dd / 128, ROWS / 128), blk, 0, stream>>>(
      xbf, wbuf, b_kv, (const float*)nullptr, f32t, (__bf16*)nullptr, ROWS, 2 * Dd, Dd, 0);
  k_pool<<<dim3(Hh, Bb), blk, 0, stream>>>(f32t, seq_ids, latent, out);
}
